// meta_model_sharp_23433341567705
// MI455X (gfx1250) — matrix operand order repaired and checked by reading, not compiled, not run on hardware
//
#include <hip/hip_runtime.h>
#include <hip/hip_bf16.h>

// ---------------------------------------------------------------------------
// Metamorphosis-style registration forward, MI455X (gfx1250, wave32, WMMA).
// conv2 (100x100, 3x3) = implicit GEMM with v_wmma_f32_16x16x32_f16.
// Per-tap weights staged into LDS by the Tensor Data Mover (TENSORcnt,
// double-buffered, overlapped with compute); B-fragments batched from LDS.
// ---------------------------------------------------------------------------

#define HH 256
#define WW 256
#define BB 8
#define HW 65536            // H*W
#define BHW 524288          // B*H*W
#define LSTEPS 5
#define SLOPE 0.01f
#define MU2_OVER_L 8.0e-5f  // 0.02*0.02/5
#define INV_L 0.2f

#define SB_STRIDE 40        // LDS row stride in f16 (32 data + 8 pad, 80B: 16B-aligned,
                            // n*20 mod 64 banks is a permutation over 16 lanes)
#define TAP_F16  (4 * 112 * 32)        // 14336 f16 = 28672 B per tap (packed)
#define TAP_LDS  (4 * 112 * SB_STRIDE) // 17920 f16 = 35840 B per tap (padded)

typedef __attribute__((ext_vector_type(16))) _Float16 v16h;
typedef __attribute__((ext_vector_type(8)))  float    v8f;
typedef __attribute__((ext_vector_type(4)))  unsigned v4u;
typedef __attribute__((ext_vector_type(8)))  int      v8i;
typedef __attribute__((ext_vector_type(4)))  int      v4i;

#if defined(__has_builtin)
#if __has_builtin(__builtin_amdgcn_tensor_load_to_lds) && \
    __has_builtin(__builtin_amdgcn_s_wait_tensorcnt)
#define USE_TDM 1
#endif
#endif
#ifndef USE_TDM
#define USE_TDM 0
#endif

__device__ __forceinline__ v16h pack16(uint4 lo, uint4 hi) {
  union { uint4 u[2]; v16h h; } w;
  w.u[0] = lo; w.u[1] = hi;
  return w.h;
}

#if USE_TDM
// TDM copy of one tap's packed weights (3584 x 8B) global -> LDS, with the
// pad feature inserting 4 DWORDs of padding every 16 DWORDs so the LDS image
// has the bank-conflict-free 80B row stride. Tracked by TENSORcnt.
// This toolchain exposes the 6-arg builtin: (v4u, v8i, v4i, v4i, v8i, i32).
__device__ __forceinline__ void tdm_copy_tap(const _Float16* gsrc, unsigned lds_off) {
  unsigned long long ga = (unsigned long long)(uintptr_t)gsrc;
  v4u g0;
  g0.x = 1u;                                        // count=1, user descriptor
  g0.y = lds_off;                                   // lds_addr (bytes)
  g0.z = (unsigned)(ga & 0xFFFFFFFFu);              // global_addr[31:0]
  g0.w = (unsigned)((ga >> 32) & 0x1FFFFFFu)        // global_addr[56:32]
         | (2u << 30);                              // type=2 ("image")
  v8i g1;
  g1[0] = (int)((3u << 16)                          // data_size=3 (8B)
                | (1u << 20)                        // pad_enable
                | (3u << 22)                        // pad_interval: 16 DWORDs
                | (3u << 25));                      // pad_amount: 4 DWORDs
  g1[1] = (int)(3584u << 16);                       // tensor_dim0[15:0] @ [63:48]
  g1[2] = (int)(1u << 16);                          // dim0 hi=0 | tensor_dim1=1
  g1[3] = (int)(3584u << 16);                       // dim1 hi=0 | tile_dim0=3584
  g1[4] = 1;                                        // tile_dim1=1, tile_dim2=0
  g1[5] = 3584;                                     // tensor_dim0_stride lo32
  g1[6] = (int)(3584u << 16);                       // stride0 hi=0 | stride1 lo16
  g1[7] = 0;                                        // stride1 hi
  v4i g2 = {0, 0, 0, 0};                            // dims 2/3 unused (2D tile)
  v4i g3 = {0, 0, 0, 0};
  v8i g4 = {0, 0, 0, 0, 0, 0, 0, 0};                // extra group (6-arg form)
  __builtin_amdgcn_tensor_load_to_lds(g0, g1, g2, g3, g4, 0);
}
#endif

__device__ __forceinline__ void fill_gauss(float* g) {
  float s = 0.f;
#pragma unroll
  for (int k = 0; k < 15; ++k) {
    float d = (float)k - 7.f;
    g[k] = __expf(-d * d * (1.f / 32.f));   // sigma = 4
    s += g[k];
  }
  float inv = 1.f / s;
#pragma unroll
  for (int k = 0; k < 15; ++k) g[k] *= inv;
}

// ------------------------------ init ---------------------------------------
__global__ void init_k(const float* __restrict__ src, const float* __restrict__ z0,
                       float* __restrict__ res0, float* __restrict__ img) {
  int id = blockIdx.x * blockDim.x + threadIdx.x;   // BHW
  res0[id] = z0[id & (HW - 1)];
  img[id]  = src[id];
}

// ------------------------------ sobel --------------------------------------
__global__ void sobel_k(const float* __restrict__ img, float* __restrict__ g) {
  int id = blockIdx.x * blockDim.x + threadIdx.x;   // BHW
  int pix = id & (HW - 1);
  int b = id >> 16;
  int y = pix >> 8, x = pix & 255;
  const float* p = img + (size_t)b * HW;
  float v[3][3];
#pragma unroll
  for (int dy = 0; dy < 3; ++dy) {
    int ys = y + dy - 1; ys = ys < 0 ? 0 : (ys > HH - 1 ? HH - 1 : ys);
#pragma unroll
    for (int dx = 0; dx < 3; ++dx) {
      int xs = x + dx - 1; xs = xs < 0 ? 0 : (xs > WW - 1 ? WW - 1 : xs);
      v[dy][dx] = p[ys * WW + xs];
    }
  }
  float gx = (-v[0][0] + v[0][2] - 2.f * v[1][0] + 2.f * v[1][2] - v[2][0] + v[2][2]) * 0.125f;
  float gy = (-v[0][0] - 2.f * v[0][1] - v[0][2] + v[2][0] + 2.f * v[2][1] + v[2][2]) * 0.125f;
  g[((size_t)(b * 2 + 0)) * HW + pix] = gx;
  g[((size_t)(b * 2 + 1)) * HW + pix] = gy;
}

// ------------------- vertical gaussian on -(r * grad) ----------------------
__global__ void blur_v_k(const float* __restrict__ res, const float* __restrict__ grads,
                         float* __restrict__ out) {
  int id = blockIdx.x * blockDim.x + threadIdx.x;   // B*2*HW
  int pix = id & (HW - 1);
  int bc = id >> 16;
  int b = bc >> 1;
  int y = pix >> 8, x = pix & 255;
  float g[15]; fill_gauss(g);
  const float* r = res + (size_t)b * HW;
  const float* gr = grads + (size_t)bc * HW;
  float s = 0.f;
#pragma unroll
  for (int k = 0; k < 15; ++k) {
    int ys = y + k - 7;
    if (ys >= 0 && ys < HH) {
      int o = ys * WW + x;
      s += g[k] * (-(r[o]) * gr[o]);          // zero pad (constant border)
    }
  }
  out[id] = s;
}

// ------------- horizontal gaussian -> field + deformation ------------------
__global__ void blur_h_k(const float* __restrict__ t2, float* __restrict__ field,
                         float* __restrict__ phi) {
  int id = blockIdx.x * blockDim.x + threadIdx.x;   // BHW
  int pix = id & (HW - 1);
  int b = id >> 16;
  int y = pix >> 8, x = pix & 255;
  float g[15]; fill_gauss(g);
  const float* p0 = t2 + ((size_t)(b * 2 + 0)) * HW + y * WW;
  const float* p1 = t2 + ((size_t)(b * 2 + 1)) * HW + y * WW;
  float s0 = 0.f, s1 = 0.f;
#pragma unroll
  for (int k = 0; k < 15; ++k) {
    int xs = x + k - 7;
    if (xs >= 0 && xs < WW) { s0 += g[k] * p0[xs]; s1 += g[k] * p1[xs]; }
  }
  size_t o = (size_t)id * 2;
  field[o]     = s0;
  field[o + 1] = s1;
  phi[o]       = (float)x - s0 * INV_L;
  phi[o + 1]   = (float)y - s1 * INV_L;
}

// ------------------------------ conv1 (3 -> 100) ---------------------------
__global__ void conv1_k(const float* __restrict__ z, const float* __restrict__ img,
                        const float* __restrict__ tgt, const float* __restrict__ w,
                        _Float16* __restrict__ a1) {
  int id = blockIdx.x * blockDim.x + threadIdx.x;   // HW*128
  int c = id & 127;
  int pix = id >> 7;
  float s = 0.f;
  if (c < 100) {
    int y = pix >> 8, x = pix & 255;
    const float* planes[3] = { z, img, tgt };
#pragma unroll
    for (int ic = 0; ic < 3; ++ic) {
      const float* p = planes[ic];
#pragma unroll
      for (int ky = 0; ky < 3; ++ky) {
        int ys = y + ky - 1;
        if (ys < 0 || ys >= HH) continue;
#pragma unroll
        for (int kx = 0; kx < 3; ++kx) {
          int xs = x + kx - 1;
          if (xs < 0 || xs >= WW) continue;
          s += p[ys * WW + xs] * w[((c * 3 + ic) * 3 + ky) * 3 + kx];
        }
      }
    }
    s = s > 0.f ? s : s * SLOPE;
  }
  a1[id] = (_Float16)s;
}

// ---------------- weight pack for conv2: wB[t][cc][n(112)][k(32)] ----------
__global__ void wprep_k(const float* __restrict__ w2, _Float16* __restrict__ wB) {
  int id = blockIdx.x * blockDim.x + threadIdx.x;   // 9*4*112*32 = 129024
  if (id >= 9 * TAP_F16) return;
  int k  = id & 31;
  int n  = (id >> 5) % 112;
  int cc = ((id >> 5) / 112) & 3;
  int t  = (id >> 5) / (112 * 4);
  int c  = cc * 32 + k;
  float v = 0.f;
  if (n < 100 && c < 100) {
    int ty = t / 3, tx = t % 3;
    v = w2[(((size_t)n * 100 + c) * 3 + ty) * 3 + tx];
  }
  wB[id] = (_Float16)v;
}

// --------------------- conv2 (100->100, 3x3) via WMMA ----------------------
// One wave = 16 output pixels x 112 output channels (7 N-tiles).
// K = 9 taps x 128 channels (4 chunks of 32) => 252 v_wmma / wave.
// TDM double-buffers per-tap weights in LDS, overlapped with compute.
__global__ __launch_bounds__(256)
void conv2_wmma_k(const _Float16* __restrict__ a1,
                  const _Float16* __restrict__ wB,
                  _Float16* __restrict__ a2) {
#if USE_TDM
  __shared__ _Float16 sB[2][TAP_LDS];            // 2 x 35840 B
#else
  __shared__ _Float16 sB1[TAP_LDS];
#endif

  const int lane  = threadIdx.x & 31;
  const int wave  = threadIdx.x >> 5;
  const int tile  = blockIdx.x * 8 + wave;   // 4096 tiles per slice
  const int y     = tile >> 4;               // 16 tiles per row
  const int x0    = (tile & 15) << 4;
  const int lhalf = lane >> 4;               // 0 | 1 (K-half select)
  const int lm    = lane & 15;

  v8f zero = {};
  v8f acc[7];
#pragma unroll
  for (int n = 0; n < 7; ++n) acc[n] = zero;

#if USE_TDM
  if (wave == 0) tdm_copy_tap(wB, (unsigned)(uintptr_t)&sB[0][0]);
#endif

  for (int t = 0; t < 9; ++t) {
#if USE_TDM
    if (wave == 0) __builtin_amdgcn_s_wait_tensorcnt(0);  // tap t landed in LDS
    __syncthreads();                                      // all: buf ready, prev reads done
    if (wave == 0 && t < 8)                               // overlap next copy w/ compute
      tdm_copy_tap(wB + (size_t)(t + 1) * TAP_F16,
                   (unsigned)(uintptr_t)&sB[(t + 1) & 1][0]);
    const _Float16* sBcur = &sB[t & 1][0];
#else
    __syncthreads();                                   // previous-tap reads done
    {
      const uint4* src = (const uint4*)(wB + (size_t)t * TAP_F16);
#pragma unroll
      for (int it = 0; it < 7; ++it) {
        int u = threadIdx.x + it * 256;                // 0..1791 uint4
        uint4 val = src[u];
        int row = u >> 2, part = u & 3;                // row = cc*112+n
        *(uint4*)(sB1 + row * SB_STRIDE + part * 8) = val;
      }
    }
    __syncthreads();
    const _Float16* sBcur = sB1;
#endif

    const int dy = t / 3 - 1, dx = t % 3 - 1;
    const int ys = y + dy;
    const int xs = x0 + dx + lm;             // this lane's A-row pixel (M = lm)
    const bool inb = (ys >= 0) & (ys < HH) & (xs >= 0) & (xs < WW);
#pragma unroll
    for (int cc = 0; cc < 4; ++cc) {
      uint4 alo = {0u, 0u, 0u, 0u}, ahi = {0u, 0u, 0u, 0u};
      if (inb) {
        const _Float16* arow = a1 + ((size_t)(ys * WW + xs) << 7);
        const int cb = (cc << 5) + (lhalf << 3);          // ISA A layout:
        alo = *(const uint4*)(arow + cb);                 //  K 0-7 / 8-15
        ahi = *(const uint4*)(arow + cb + 16);            //  K 16-23 / 24-31
      }
      const v16h A = pack16(alo, ahi);
      const _Float16* wb = sBcur + (cc * 112 + lm) * SB_STRIDE + lhalf * 8;
      v16h Bv[7];                             // batch all 7 B-fragments first
#pragma unroll
      for (int nt = 0; nt < 7; ++nt) {
        const _Float16* wp = wb + nt * (16 * SB_STRIDE);
        Bv[nt] = pack16(*(const uint4*)(wp), *(const uint4*)(wp + 16));
      }
#pragma unroll
      for (int nt = 0; nt < 7; ++nt)
        acc[nt] = __builtin_amdgcn_wmma_f32_16x16x32_f16(
            false, A, false, Bv[nt], (short)0, acc[nt], false, false);
    }
  }

  // Epilogue: leaky ReLU + f16 store, C/D layout: VGPR r -> M=r (+8 hi lanes)
  const int pbase = y * WW + x0 + lhalf * 8;
#pragma unroll
  for (int nt = 0; nt < 7; ++nt) {
#pragma unroll
    for (int r = 0; r < 8; ++r) {
      float v = acc[nt][r];
      v = v > 0.f ? v : v * SLOPE;
      a2[(((size_t)(pbase + r)) << 7) + nt * 16 + lm] = (_Float16)v;
    }
  }
}

// ------------------- conv3 (100 -> 1) + residual update --------------------
__global__ void conv3_k(const _Float16* __restrict__ a2, const float* __restrict__ w3,
                        const float* __restrict__ rold, float* __restrict__ rnew) {
  int pix = blockIdx.x * blockDim.x + threadIdx.x;   // HW
  int y = pix >> 8, x = pix & 255;
  float s = 0.f;
#pragma unroll
  for (int ky = 0; ky < 3; ++ky) {
    int ys = y + ky - 1;
    if (ys < 0 || ys >= HH) continue;
#pragma unroll
    for (int kx = 0; kx < 3; ++kx) {
      int xs = x + kx - 1;
      if (xs < 0 || xs >= WW) continue;
      const _Float16* ar = a2 + ((size_t)(ys * WW + xs) << 7);
      const float* wt = w3 + ky * 3 + kx;            // stride 9 over channels
      float ps = 0.f;
      for (int c0 = 0; c0 < 96; c0 += 8) {           // 8-wide f16 loads
        union { uint4 u; _Float16 h[8]; } q;
        q.u = *(const uint4*)(ar + c0);
#pragma unroll
        for (int j = 0; j < 8; ++j) ps += (float)q.h[j] * wt[(c0 + j) * 9];
      }
#pragma unroll
      for (int c = 96; c < 100; ++c) ps += (float)ar[c] * wt[c * 9];
      s += ps;
    }
  }
  rnew[pix] = rold[pix] - s * INV_L;                  // r_{i+1} = r_i - f/L
}

// ---------------------- bilinear samplers (zeros pad) ----------------------
__device__ __forceinline__ float bilerp1(const float* __restrict__ p, float gx, float gy) {
  float fx0 = floorf(gx), fy0 = floorf(gy);
  int x0 = (int)fx0, y0 = (int)fy0;
  float ax = gx - fx0, ay = gy - fy0;
  float acc = 0.f;
#pragma unroll
  for (int dy = 0; dy < 2; ++dy)
#pragma unroll
    for (int dx = 0; dx < 2; ++dx) {
      int xi = x0 + dx, yi = y0 + dy;
      if (xi < 0 || xi > WW - 1 || yi < 0 || yi > HH - 1) continue;
      float w = (dx ? ax : 1.f - ax) * (dy ? ay : 1.f - ay);
      acc += p[yi * WW + xi] * w;
    }
  return acc;
}

__device__ __forceinline__ void bilerp2(const float* __restrict__ p, float gx, float gy,
                                        float& o0, float& o1) {
  float fx0 = floorf(gx), fy0 = floorf(gy);
  int x0 = (int)fx0, y0 = (int)fy0;
  float ax = gx - fx0, ay = gy - fy0;
  o0 = 0.f; o1 = 0.f;
#pragma unroll
  for (int dy = 0; dy < 2; ++dy)
#pragma unroll
    for (int dx = 0; dx < 2; ++dx) {
      int xi = x0 + dx, yi = y0 + dy;
      if (xi < 0 || xi > WW - 1 || yi < 0 || yi > HH - 1) continue;
      float w = (dx ? ax : 1.f - ax) * (dy ? ay : 1.f - ay);
      const float* q = p + ((size_t)(yi * WW + xi)) * 2;
      o0 += q[0] * w; o1 += q[1] * w;
    }
}

// -------------------- warp a [B,H,W,2] phi by deformation ------------------
__global__ void warp2_k(const float* __restrict__ phik, const float* __restrict__ defo,
                        float* __restrict__ out) {
  int id = blockIdx.x * blockDim.x + threadIdx.x;   // BHW
  int b = id >> 16;
  const float* pk = phik + ((size_t)b * HW) * 2;
  float gx = defo[(size_t)id * 2], gy = defo[(size_t)id * 2 + 1];
  float s0, s1;
  bilerp2(pk, gx, gy, s0, s1);
  out[(size_t)id * 2] = s0;
  out[(size_t)id * 2 + 1] = s1;
}

// -------- compose: image = warp(src, phi0) + (r_{i+1} + sum warps)*mu^2/L --
__global__ void compose_k(const float* __restrict__ res_base, const float* __restrict__ phi_base,
                          const float* __restrict__ source, int i, float* __restrict__ image) {
  int id = blockIdx.x * blockDim.x + threadIdx.x;   // BHW
  int b = id >> 16;
  float acc = res_base[(size_t)(i + 1) * BHW + id];
  for (int k = 1; k <= i; ++k) {
    const float* ph = phi_base + (size_t)k * (BHW * 2) + (size_t)id * 2;
    acc += bilerp1(res_base + (size_t)k * BHW + (size_t)b * HW, ph[0], ph[1]);
  }
  const float* ph0 = phi_base + (size_t)id * 2;
  image[id] = bilerp1(source + (size_t)b * HW, ph0[0], ph0[1]) + acc * MU2_OVER_L;
}

// ---------------------------------------------------------------------------
extern "C" void kernel_launch(void* const* d_in, const int* in_sizes, int n_in,
                              void* d_out, int out_size, void* d_ws, size_t ws_size,
                              hipStream_t stream) {
  const float* source = (const float*)d_in[0];
  const float* target = (const float*)d_in[1];
  // d_in[2] (source_seg) unused by the reference forward
  const float* z0 = (const float*)d_in[3];
  const float* w1 = (const float*)d_in[4];   // [5,100,3,3,3]
  const float* w2 = (const float*)d_in[5];   // [5,100,100,3,3]
  const float* w3 = (const float*)d_in[6];   // [5,1,100,3,3]

  float* out        = (float*)d_out;
  float* out_image  = out;                       // [8,1,256,256]
  float* out_fields = out + 524288;              // [5,8,256,256,2]
  float* out_res    = out + 5767168;             // [6,8,1,256,256]
  float* out_grads  = out + 8912896;             // [5,8,2,256,256]

  char* ws = (char*)d_ws;
  float*    image_ws = (float*)ws;               ws += (size_t)BHW * 4;              // 2 MB
  float*    tmp2     = (float*)ws;               ws += (size_t)2 * BHW * 4;          // 4 MB
  float*    phi      = (float*)ws;               ws += (size_t)LSTEPS * BHW * 2 * 4; // 20 MB
  float*    scr      = (float*)ws;               ws += (size_t)BHW * 2 * 4;          // 4 MB
  _Float16* a1       = (_Float16*)ws;            ws += (size_t)HW * 128 * 2;         // 16.8 MB
  _Float16* a2       = (_Float16*)ws;            ws += (size_t)HW * 128 * 2;         // 16.8 MB
  _Float16* wB       = (_Float16*)ws;            ws += (size_t)9 * TAP_F16 * 2;

  init_k<<<BHW / 256, 256, 0, stream>>>(source, z0, out_res /*slab 0*/, image_ws);

  for (int i = 0; i < LSTEPS; ++i) {
    float* grads_i = out_grads + (size_t)i * (2 * BHW);
    float* res_i   = out_res + (size_t)i * BHW;
    float* res_ip1 = out_res + (size_t)(i + 1) * BHW;
    float* field_i = out_fields + (size_t)i * (BHW * 2);
    float* phi_i   = phi + (size_t)i * (BHW * 2);

    sobel_k <<<BHW / 256, 256, 0, stream>>>(image_ws, grads_i);
    blur_v_k<<<2 * BHW / 256, 256, 0, stream>>>(res_i, grads_i, tmp2);
    blur_h_k<<<BHW / 256, 256, 0, stream>>>(tmp2, field_i, phi_i);

    wprep_k<<<(9 * TAP_F16) / 256, 256, 0, stream>>>(w2 + (size_t)i * 90000, wB);

    for (int b = 0; b < BB; ++b) {
      conv1_k<<<(HW * 128) / 256, 256, 0, stream>>>(
          res_i + (size_t)b * HW, image_ws + (size_t)b * HW, target + (size_t)b * HW,
          w1 + (size_t)i * 2700, a1);
      conv2_wmma_k<<<512, 256, 0, stream>>>(a1, wB, a2);
      conv3_k<<<HW / 256, 256, 0, stream>>>(
          a2, w3 + (size_t)i * 900, res_i + (size_t)b * HW, res_ip1 + (size_t)b * HW);
    }

    // warp existing phi entries by this step's deformation (phi_i)
    for (int k = 0; k < i; ++k) {
      float* phi_k = phi + (size_t)k * (BHW * 2);
      warp2_k<<<BHW / 256, 256, 0, stream>>>(phi_k, phi_i, scr);
      (void)hipMemcpyAsync(phi_k, scr, (size_t)BHW * 2 * 4, hipMemcpyDeviceToDevice, stream);
    }

    compose_k<<<BHW / 256, 256, 0, stream>>>(out_res, phi, source, i, image_ws);
  }

  (void)hipMemcpyAsync(out_image, image_ws, (size_t)BHW * 4, hipMemcpyDeviceToDevice, stream);
}
